// SparseSAE_4990751998027
// MI455X (gfx1250) — matrix operand order repaired and checked by reading, not compiled, not run on hardware
//
#include <hip/hip_runtime.h>

// SAE forward for MI455X (gfx1250):
//   z_pre = x @ enc_w^T + enc_b   (8192x8192, f16 WMMA, fp32 accumulate)
//   z     = scatter(relu(top20(z_pre)))
//   x_hat = z @ dec_w^T + dec_b   (sparse decode, 20 gathered rows of dec_w^T)
//
// d_out = [ z : 8192*8192 f32 | x_hat : 8192*1024 f32 ]
// d_ws  = [ x_f16 : 16MB | enc_w_f16 : 16MB | dec_wT_f32 : 32MB ]  (64MB total)

#define TOPK 20
constexpr int Cc = 1024;   // model dim
constexpr int Kk = 8192;   // dictionary size
constexpr int Tt = 8192;   // tokens = B*N = 32*256

typedef _Float16 v8h  __attribute__((ext_vector_type(8)));
typedef _Float16 v16h __attribute__((ext_vector_type(16)));
typedef float    v8f  __attribute__((ext_vector_type(8)));
typedef _Float16 h4   __attribute__((ext_vector_type(4)));
typedef int      v4i  __attribute__((ext_vector_type(4)));

typedef __attribute__((address_space(1))) v4i* gv4i_p;  // global 128-bit ptr
typedef __attribute__((address_space(3))) v4i* lv4i_p;  // LDS    128-bit ptr

// ---------------- CDNA5 async global->LDS copy helpers ----------------------
__device__ __forceinline__ void async_cp_b128(const void* gsrc, void* ldst) {
#if __has_builtin(__builtin_amdgcn_global_load_async_to_lds_b128)
  __builtin_amdgcn_global_load_async_to_lds_b128(
      (gv4i_p)gsrc, (lv4i_p)ldst, 0, 0);
#else
  unsigned loff = (unsigned)(unsigned long long)
      (__attribute__((address_space(3))) void*)ldst;
  asm volatile("global_load_async_to_lds_b128 %0, %1, off"
               :: "v"(loff), "v"((unsigned long long)gsrc) : "memory");
#endif
}

__device__ __forceinline__ void wait_async0() {
#if __has_builtin(__builtin_amdgcn_s_wait_asynccnt)
  __builtin_amdgcn_s_wait_asynccnt(0);
#else
  asm volatile("s_wait_asynccnt 0" ::: "memory");
#endif
}

// ---------------- K0a: fp32 -> f16 conversion (4 elems/thread) --------------
__global__ void cvt_f16(const float* __restrict__ s, _Float16* __restrict__ d,
                        int n4) {
  int i = blockIdx.x * 256 + threadIdx.x;
  if (i >= n4) return;
  float4 v = ((const float4*)s)[i];
  h4 o;
  o.x = (_Float16)v.x; o.y = (_Float16)v.y;
  o.z = (_Float16)v.z; o.w = (_Float16)v.w;
  ((h4*)d)[i] = o;
}

// ---------------- K0b: transpose dec_w (C x K) -> dec_wT (K x C) ------------
__global__ void transpose_cw(const float* __restrict__ src,
                             float* __restrict__ dst) {
  __shared__ float tile[32][33];
  const int k0 = blockIdx.x * 32, c0 = blockIdx.y * 32;
  const int tx = threadIdx.x, ty = threadIdx.y;   // 32 x 8 threads
#pragma unroll
  for (int j = 0; j < 32; j += 8)
    tile[ty + j][tx] = src[(size_t)(c0 + ty + j) * Kk + (k0 + tx)];
  __syncthreads();
#pragma unroll
  for (int j = 0; j < 32; j += 8)
    dst[(size_t)(k0 + ty + j) * Cc + (c0 + tx)] = tile[tx][ty + j];
}

// ---------------- K1: encode GEMM, f16 WMMA, 128x128 tile per WG ------------
constexpr int LDA = 40;  // padded LDS row stride in halves (32 data + 8 pad)

__global__ __launch_bounds__(256)
void enc_gemm(const _Float16* __restrict__ xh, const _Float16* __restrict__ wh,
              const float* __restrict__ encb, float* __restrict__ zout) {
  __shared__ alignas(16) _Float16 lA[2][128 * LDA];  // 10 KB x2
  __shared__ alignas(16) _Float16 lB[2][128 * LDA];  // 10 KB x2

  const int t    = threadIdx.x;
  const int wave = t >> 5;
  const int lane = t & 31;
  const int lm   = lane & 15;
  const int hi   = lane >> 4;          // half-wave select
  const int waveM = wave & 1;          // 2 groups of 64 rows
  const int waveN = wave >> 1;         // 4 groups of 32 cols
  const int rowBase = blockIdx.y * 128;
  const int colBase = blockIdx.x * 128;

  // staging: 128 rows x 32 halves per tile, 2 threads per row, 32B each
  const int srow  = t >> 1;
  const int shalf = (t & 1) * 16;
  const _Float16* gA = xh + (size_t)(rowBase + srow) * Cc + shalf;
  const _Float16* gB = wh + (size_t)(colBase + srow) * Cc + shalf;

  auto stage = [&](int buf, int kb) {
    const _Float16* a = gA + kb;
    const _Float16* b = gB + kb;
    _Float16* da = &lA[buf][srow * LDA + shalf];
    _Float16* db = &lB[buf][srow * LDA + shalf];
    async_cp_b128(a,     da);
    async_cp_b128(a + 8, da + 8);
    async_cp_b128(b,     db);
    async_cp_b128(b + 8, db + 8);
  };

  v8f acc[4][2] = {};
  stage(0, 0);

  for (int ks = 0; ks < Cc / 32; ++ks) {
    const int cur = ks & 1;
    wait_async0();
    __syncthreads();
    if (ks + 1 < Cc / 32) stage(cur ^ 1, (ks + 1) * 32);

    // B fragments: 32x16, lane holds 16 contiguous K at K = hi*16
    v16h bf[2];
#pragma unroll
    for (int cg = 0; cg < 2; ++cg) {
      const _Float16* p =
          &lB[cur][(waveN * 32 + cg * 16 + lm) * LDA + hi * 16];
      ((v8h*)&bf[cg])[0] = *(const v8h*)p;
      ((v8h*)&bf[cg])[1] = *(const v8h*)(p + 8);
    }
#pragma unroll
    for (int rg = 0; rg < 4; ++rg) {
      // A fragment: 16x32, lane holds 16 contiguous K at K = hi*16
      const _Float16* p = &lA[cur][(waveM * 64 + rg * 16 + lm) * LDA];
      v16h af;
      ((v8h*)&af)[0] = *(const v8h*)(p + hi * 16);
      ((v8h*)&af)[1] = *(const v8h*)(p + hi * 16 + 8);
#pragma unroll
      for (int cg = 0; cg < 2; ++cg)
        acc[rg][cg] = __builtin_amdgcn_wmma_f32_16x16x32_f16(
            false, af, false, bf[cg], (short)0, acc[rg][cg], false, false);
    }
  }

  // epilogue: z_pre = acc + enc_b[col]
#pragma unroll
  for (int cg = 0; cg < 2; ++cg) {
    const int col = colBase + waveN * 32 + cg * 16 + lm;
    const float bias = encb[col];
#pragma unroll
    for (int rg = 0; rg < 4; ++rg) {
      const int row0 = rowBase + waveM * 64 + rg * 16 + hi * 8;
#pragma unroll
      for (int r = 0; r < 8; ++r)
        zout[(size_t)(row0 + r) * Kk + col] = acc[rg][cg][r] + bias;
    }
  }
}

// ---------------- K2: per-row top-20 + ReLU + scatter + sparse decode -------
__global__ __launch_bounds__(256)
void topk_decode(float* __restrict__ z, const float* __restrict__ dwt,
                 const float* __restrict__ decb, float* __restrict__ xhat) {
  __shared__ float rowv[Kk];      // 32 KB row cache
  __shared__ float rv[256];
  __shared__ int   ri[256];
  __shared__ float selv[TOPK];
  __shared__ int   seli[TOPK];

  const int t = threadIdx.x;
  const int row = blockIdx.x;
  float* zrow = z + (size_t)row * Kk;

#pragma unroll
  for (int j = 0; j < Kk / 256; ++j) rowv[j * 256 + t] = zrow[j * 256 + t];
  __syncthreads();

  for (int it = 0; it < TOPK; ++it) {
    float m = -3.4e38f; int mi = t;
    for (int j = 0; j < Kk / 256; ++j) {
      float v = rowv[j * 256 + t];
      if (v > m) { m = v; mi = j * 256 + t; }
    }
    rv[t] = m; ri[t] = mi;
    __syncthreads();
    for (int s = 128; s > 0; s >>= 1) {
      if (t < s && rv[t + s] > rv[t]) { rv[t] = rv[t + s]; ri[t] = ri[t + s]; }
      __syncthreads();
    }
    if (t == 0) {
      const int ix = ri[0];
      selv[it] = rv[0] > 0.f ? rv[0] : 0.f;   // ReLU on kept values
      seli[it] = ix;
      rowv[ix] = -3.4e38f;
    }
    __syncthreads();
  }

  // rewrite z row in place: zeros + scattered kept values
  float4* z4 = (float4*)zrow;
  const float4 zero4 = make_float4(0.f, 0.f, 0.f, 0.f);
#pragma unroll
  for (int j = 0; j < Kk / 4 / 256; ++j) z4[j * 256 + t] = zero4;
  __syncthreads();
  if (t < TOPK) zrow[seli[t]] = selv[t];

  // sparse decode: x_hat[c] = dec_b[c] + sum_j selv[j] * dec_wT[seli[j]][c]
#pragma unroll
  for (int i = 0; i < Cc / 256; ++i) {
    const int c = i * 256 + t;
    float acc = decb[c];
#pragma unroll
    for (int j = 0; j < TOPK; ++j)
      acc += selv[j] * dwt[(size_t)seli[j] * Cc + c];
    xhat[(size_t)row * Cc + c] = acc;
  }
}

// ---------------- launch ----------------------------------------------------
extern "C" void kernel_launch(void* const* d_in, const int* in_sizes, int n_in,
                              void* d_out, int out_size, void* d_ws,
                              size_t ws_size, hipStream_t stream) {
  const float* x  = (const float*)d_in[0];   // (B,N,C)
  const float* ew = (const float*)d_in[1];   // (K,C)
  const float* eb = (const float*)d_in[2];   // (K,)
  const float* dw = (const float*)d_in[3];   // (C,K)
  const float* db = (const float*)d_in[4];   // (C,)

  float* z    = (float*)d_out;               // (T,K)
  float* xhat = z + (size_t)Tt * Kk;         // (T,C)

  _Float16* xh16 = (_Float16*)d_ws;                    // 16 MB
  _Float16* wh16 = xh16 + (size_t)Tt * Cc;             // 16 MB
  float*    dwt  = (float*)(wh16 + (size_t)Kk * Cc);   // 32 MB

  cvt_f16<<<(Tt * Cc / 4) / 256, 256, 0, stream>>>(x, xh16, Tt * Cc / 4);
  cvt_f16<<<(Kk * Cc / 4) / 256, 256, 0, stream>>>(ew, wh16, Kk * Cc / 4);
  transpose_cw<<<dim3(Kk / 32, Cc / 32), dim3(32, 8), 0, stream>>>(dw, dwt);
  enc_gemm<<<dim3(Kk / 128, Tt / 128), 256, 0, stream>>>(xh16, wh16, eb, z);
  topk_decode<<<Tt, 256, 0, stream>>>(z, dwt, db, xhat);
}
